// GCBlock3_558345748932
// MI455X (gfx1250) — matrix operand order repaired and checked by reading, not compiled, not run on hardware
//
#include <hip/hip_runtime.h>

// ---------------------------------------------------------------------------
// GCBlock3 fused edge/node kernels for MI455X (gfx1250, wave32, WMMA f16->f32)
// N=10000 nodes, E=320000 edges, F=64 features, NB=8 basis
// ---------------------------------------------------------------------------

typedef __attribute__((ext_vector_type(16))) _Float16 v16h;
typedef __attribute__((ext_vector_type(8)))  _Float16 v8h;
typedef __attribute__((ext_vector_type(8)))  float    v8f;

#define FDIM 64

// Native CDNA5 transcendental tanh (TRANS32 op, co-executes with WMMA pipe).
// Avoids the ~50-VALU branchy libm expansion of tanhf().
__device__ __forceinline__ float fast_tanh(float x) {
#if __has_builtin(__builtin_amdgcn_tanhf)
  return __builtin_amdgcn_tanhf(x);
#else
  float r;
  // TRANS op: pad with v_nops per ISA hazard rule (compiler can't see asm deps)
  asm volatile("v_tanh_f32 %0, %1\n\tv_nop\n\tv_nop" : "=v"(r) : "v"(x));
  return r;
#endif
}

__device__ __forceinline__ v8f wmma_f16(v16h a, v16h b, v8f c) {
  // D = A(16x32 f16) * B(32x16 f16) + C(16x16 f32)
  return __builtin_amdgcn_wmma_f32_16x16x32_f16(false, a, false, b,
                                                (short)0, c, false, false);
}

// A-fragment: 16x32 f16 tile from row-major LDS buffer (stride = 64 halves).
// ISA layout: lane m=l&15; lanes 0-15 hold K=[k0+0..7]+[k0+16..23],
// lanes 16-31 hold K=[k0+8..15]+[k0+24..31]  -> two contiguous v8h chunks.
__device__ __forceinline__ v16h load_a_frag(const _Float16* base, int k0, int lane) {
  int m  = lane & 15;
  int b0 = k0 + ((lane >> 4) << 3);
  union { v16h v; v8h h[2]; } u;
  u.h[0] = *(const v8h*)(base + m * FDIM + b0);
  u.h[1] = *(const v8h*)(base + m * FDIM + b0 + 16);
  return u.v;
}

// B-fragment: 32x16 f16 tile taken from a transposed weight W^T stored as
// (Ncols x 64K) row-major in global (L2-resident). ISA layout: lane n=l&15;
// lanes 0-15 hold K=[k0+0..7]+[k0+16..23], lanes 16-31 hold K=[k0+8..15]+[k0+24..31].
__device__ __forceinline__ v16h load_b_fragT(const _Float16* wT, int n0, int k0, int lane) {
  int n  = n0 + (lane & 15);
  int kb = k0 + ((lane >> 4) << 3);
  union { v16h v; v8h h[2]; } u;
  u.h[0] = *(const v8h*)(wT + n * FDIM + kb);
  u.h[1] = *(const v8h*)(wT + n * FDIM + kb + 16);
  return u.v;
}

// ---------------------------------------------------------------------------
// Prep kernels
// ---------------------------------------------------------------------------
__global__ void zero_f32_kernel(float* p, int n) {
  int i = blockIdx.x * 256 + threadIdx.x;
  if (i < n) p[i] = 0.0f;
}

// wT[n*K + k] = w[k*Ncol + n]  (f32 -> f16)
__global__ void transpose_to_f16_kernel(const float* __restrict__ w,
                                        _Float16* __restrict__ wT,
                                        int K, int Ncol) {
  int i = blockIdx.x * 256 + threadIdx.x;
  if (i < K * Ncol) {
    int n = i / K, k = i % K;
    wT[i] = (_Float16)w[k * Ncol + n];
  }
}

// ---------------------------------------------------------------------------
// Edge kernel: 4 waves/block, one 16-edge tile per wave.
// Computes i1 (output 3), ix (output 4), and f32 atomic segment-sums into
// p1acc (N x 64) and p3acc (N x 3 x 64).
// ---------------------------------------------------------------------------
__global__ __launch_bounds__(128) void edge_kernel(
    const float* __restrict__ p1, const float* __restrict__ p3,
    const float* __restrict__ basis, const float* __restrict__ diff,
    const int* __restrict__ pair_i, const int* __restrict__ pair_j,
    const _Float16* __restrict__ w0T,   // 64 x 64 (pi_W0^T)
    const _Float16* __restrict__ w1T,   // 512 x 64 (pi_W1^T)
    const _Float16* __restrict__ iiT,   // 64 x 64 (ii_W^T)
    const _Float16* __restrict__ epiT,  // 64 x 64 (epi_W^T)
    float* __restrict__ out_i1, float* __restrict__ out_ix,
    float* __restrict__ p1acc, float* __restrict__ p3acc, int E) {
  __shared__ alignas(16) _Float16 sA[4][16 * FDIM];      // s = p1[i]+p1[j] tile
  __shared__ alignas(16) _Float16 tA[4][16 * FDIM];      // tanh(s@W0) tile
  __shared__ alignas(16) _Float16 i1preA[4][16 * FDIM];  // basis-contracted h
  __shared__ alignas(16) _Float16 qA[4][16 * FDIM];      // p3[i]+p3[j] slice
  __shared__ float sBasis[4][16 * 8];
  __shared__ float sDiff[4][16 * 3];
  __shared__ int   sIdxI[4][16];
  __shared__ int   sIdxJ[4][16];

  const int wave = threadIdx.x >> 5;
  const int lane = threadIdx.x & 31;
  const int tile = blockIdx.x * 4 + wave;
  const bool active = (tile * 16 + 16) <= E;
  const int e0 = active ? tile * 16 : 0;

  _Float16* mySA  = sA[wave];
  _Float16* myTA  = tA[wave];
  _Float16* myI1P = i1preA[wave];
  _Float16* myQA  = qA[wave];

  if (lane < 16) {
    sIdxI[wave][lane] = pair_i[e0 + lane];
    sIdxJ[wave][lane] = pair_j[e0 + lane];
  }
  __syncthreads();

  // Gather: s = p1[i] + p1[j] (f16), basis, diff
  for (int r = 0; r < 4; ++r) {
    int t = r * 32 + lane;
    sBasis[wave][t] = basis[e0 * 8 + t];
  }
  for (int t = lane; t < 48; t += 32) sDiff[wave][t] = diff[e0 * 3 + t];
  for (int r = 0; r < 32; ++r) {
    int t = r * 32 + lane;
    int m = t >> 6, f = t & 63;
    float v = p1[sIdxI[wave][m] * FDIM + f] + p1[sIdxJ[wave][m] * FDIM + f];
    mySA[m * FDIM + f] = (_Float16)v;
  }
  __syncthreads();

  // t = tanh(s @ pi_W0)  : 4 col tiles x (K=64 -> 2 wmma)
  const int nlo = lane & 15;
  const int mb  = (lane >> 4) << 3;  // C-fragment row base for this lane
  for (int n0 = 0; n0 < FDIM; n0 += 16) {
    v8f acc = {};
    acc = wmma_f16(load_a_frag(mySA, 0,  lane), load_b_fragT(w0T, n0, 0,  lane), acc);
    acc = wmma_f16(load_a_frag(mySA, 32, lane), load_b_fragT(w0T, n0, 32, lane), acc);
    int n = n0 + nlo;
#pragma unroll
    for (int r = 0; r < 8; ++r) myTA[(mb + r) * FDIM + n] = (_Float16)fast_tanh(acc[r]);
  }
  __syncthreads();

  // h = tanh(t @ pi_W1) (E x 512), fused with basis contraction over b:
  // i1pre[e][f] = sum_b h[e][f*8+b] * basis[e][b]
  for (int c0 = 0; c0 < 512; c0 += 16) {
    v8f acc = {};
    acc = wmma_f16(load_a_frag(myTA, 0,  lane), load_b_fragT(w1T, c0, 0,  lane), acc);
    acc = wmma_f16(load_a_frag(myTA, 32, lane), load_b_fragT(w1T, c0, 32, lane), acc);
    int c = c0 + nlo;
    int b = c & 7, fcol = c >> 3;
    float vals[8];
#pragma unroll
    for (int r = 0; r < 8; ++r)
      vals[r] = fast_tanh(acc[r]) * sBasis[wave][(mb + r) * 8 + b];
    // reduce over b = groups of 8 consecutive lanes
#pragma unroll
    for (int s = 1; s < 8; s <<= 1)
#pragma unroll
      for (int r = 0; r < 8; ++r) vals[r] += __shfl_xor(vals[r], s, 32);
    if ((lane & 7) == 0) {
#pragma unroll
      for (int r = 0; r < 8; ++r) myI1P[(mb + r) * FDIM + fcol] = (_Float16)vals[r];
    }
  }
  __syncthreads();

  // i1 = tanh(i1pre @ ii_W); keep fragments in registers, emit output + atomics
  v8f i1frag[4];
  for (int j = 0; j < 4; ++j) {
    v8f acc = {};
    acc = wmma_f16(load_a_frag(myI1P, 0,  lane), load_b_fragT(iiT, j * 16, 0,  lane), acc);
    acc = wmma_f16(load_a_frag(myI1P, 32, lane), load_b_fragT(iiT, j * 16, 32, lane), acc);
#pragma unroll
    for (int r = 0; r < 8; ++r) acc[r] = fast_tanh(acc[r]);
    i1frag[j] = acc;
    int n = j * 16 + nlo;
    if (active) {
#pragma unroll
      for (int r = 0; r < 8; ++r) {
        int m = mb + r;
        out_i1[(e0 + m) * FDIM + n] = acc[r];
        atomicAdd(&p1acc[sIdxI[wave][m] * FDIM + n], acc[r]);
      }
    }
  }

  // ix[e][x][f] = ((p3[i]+p3[j]) @ epi_W + diff[e][x]) * i1[e][f]
  for (int x = 0; x < 3; ++x) {
    __syncthreads();
    for (int r = 0; r < 32; ++r) {
      int t = r * 32 + lane;
      int m = t >> 6, f = t & 63;
      float v = p3[(sIdxI[wave][m] * 3 + x) * FDIM + f] +
                p3[(sIdxJ[wave][m] * 3 + x) * FDIM + f];
      myQA[m * FDIM + f] = (_Float16)v;
    }
    __syncthreads();
    for (int j = 0; j < 4; ++j) {
      v8f acc = {};
      acc = wmma_f16(load_a_frag(myQA, 0,  lane), load_b_fragT(epiT, j * 16, 0,  lane), acc);
      acc = wmma_f16(load_a_frag(myQA, 32, lane), load_b_fragT(epiT, j * 16, 32, lane), acc);
      int n = j * 16 + nlo;
      if (active) {
#pragma unroll
        for (int r = 0; r < 8; ++r) {
          int m = mb + r;
          float val = (acc[r] + sDiff[wave][m * 3 + x]) * i1frag[j][r];
          out_ix[((e0 + m) * 3 + x) * FDIM + n] = val;
          atomicAdd(&p3acc[(sIdxI[wave][m] * 3 + x) * FDIM + n], val);
        }
      }
    }
  }
}

// ---------------------------------------------------------------------------
// Node kernel: 2 waves/block, one 16-node tile per wave.
// p1n = tanh(p1acc @ ipp_W + b); p3n = p3acc @ epp_W; dot = sum_x p3n^2;
// px = [p1n; dot] @ gpp_W + b; p1t1 = px0; p3t1 = p3n * px1.
// ---------------------------------------------------------------------------
__global__ __launch_bounds__(64) void node_kernel(
    const float* __restrict__ p1acc, const float* __restrict__ p3acc,
    const _Float16* __restrict__ ippT, const float* __restrict__ ipp_b,
    const _Float16* __restrict__ eppT,
    const _Float16* __restrict__ gppT, const float* __restrict__ gpp_b,
    float* __restrict__ out_p1t1, float* __restrict__ out_p3t1, int N) {
  __shared__ alignas(16) _Float16 aA[2][16 * FDIM];
  __shared__ alignas(16) _Float16 p1nA[2][16 * FDIM];
  __shared__ alignas(16) _Float16 dotA[2][16 * FDIM];
  __shared__ float p3nS[2][3][16 * FDIM];

  const int wave = threadIdx.x >> 5;
  const int lane = threadIdx.x & 31;
  const int tile = blockIdx.x * 2 + wave;
  const bool active = (tile * 16 + 16) <= N;
  const int v0 = active ? tile * 16 : 0;

  _Float16* myA = aA[wave];
  const int nlo = lane & 15;
  const int mb  = (lane >> 4) << 3;

  // p1n = tanh(p1acc @ ipp_W + b)
  for (int r = 0; r < 32; ++r) {
    int t = r * 32 + lane;
    int m = t >> 6, f = t & 63;
    myA[m * FDIM + f] = (_Float16)p1acc[(v0 + m) * FDIM + f];
  }
  __syncthreads();
  for (int j = 0; j < 4; ++j) {
    v8f acc = {};
    acc = wmma_f16(load_a_frag(myA, 0,  lane), load_b_fragT(ippT, j * 16, 0,  lane), acc);
    acc = wmma_f16(load_a_frag(myA, 32, lane), load_b_fragT(ippT, j * 16, 32, lane), acc);
    int n = j * 16 + nlo;
    float bn = ipp_b[n];
#pragma unroll
    for (int r = 0; r < 8; ++r)
      p1nA[wave][(mb + r) * FDIM + n] = (_Float16)fast_tanh(acc[r] + bn);
  }

  // p3n[x] = p3acc[x] @ epp_W ; dot += p3n^2 (fragment-aligned accumulation)
  v8f dotf[4] = {};
  for (int x = 0; x < 3; ++x) {
    __syncthreads();
    for (int r = 0; r < 32; ++r) {
      int t = r * 32 + lane;
      int m = t >> 6, f = t & 63;
      myA[m * FDIM + f] = (_Float16)p3acc[((v0 + m) * 3 + x) * FDIM + f];
    }
    __syncthreads();
    for (int j = 0; j < 4; ++j) {
      v8f acc = {};
      acc = wmma_f16(load_a_frag(myA, 0,  lane), load_b_fragT(eppT, j * 16, 0,  lane), acc);
      acc = wmma_f16(load_a_frag(myA, 32, lane), load_b_fragT(eppT, j * 16, 32, lane), acc);
      int n = j * 16 + nlo;
#pragma unroll
      for (int r = 0; r < 8; ++r) {
        p3nS[wave][x][(mb + r) * FDIM + n] = acc[r];
        dotf[j][r] += acc[r] * acc[r];
      }
    }
  }
  for (int j = 0; j < 4; ++j) {
    int n = j * 16 + nlo;
#pragma unroll
    for (int r = 0; r < 8; ++r)
      dotA[wave][(mb + r) * FDIM + n] = (_Float16)dotf[j][r];
  }
  __syncthreads();

  // px0 = p1n @ gpp_W + b  -> p1t1
  for (int j = 0; j < 4; ++j) {
    v8f acc = {};
    acc = wmma_f16(load_a_frag(p1nA[wave], 0,  lane), load_b_fragT(gppT, j * 16, 0,  lane), acc);
    acc = wmma_f16(load_a_frag(p1nA[wave], 32, lane), load_b_fragT(gppT, j * 16, 32, lane), acc);
    int n = j * 16 + nlo;
    float bn = gpp_b[n];
    if (active) {
#pragma unroll
      for (int r = 0; r < 8; ++r) out_p1t1[(v0 + mb + r) * FDIM + n] = acc[r] + bn;
    }
  }
  // px1 = dot @ gpp_W + b  -> p3t1 = p3n * px1
  for (int j = 0; j < 4; ++j) {
    v8f acc = {};
    acc = wmma_f16(load_a_frag(dotA[wave], 0,  lane), load_b_fragT(gppT, j * 16, 0,  lane), acc);
    acc = wmma_f16(load_a_frag(dotA[wave], 32, lane), load_b_fragT(gppT, j * 16, 32, lane), acc);
    int n = j * 16 + nlo;
    float bn = gpp_b[n];
    if (active) {
#pragma unroll
      for (int r = 0; r < 8; ++r) {
        float px1 = acc[r] + bn;
        int m = mb + r;
        for (int x = 0; x < 3; ++x)
          out_p3t1[((v0 + m) * 3 + x) * FDIM + n] = p3nS[wave][x][m * FDIM + n] * px1;
      }
    }
  }
}

// ---------------------------------------------------------------------------
// Host launcher
// ---------------------------------------------------------------------------
extern "C" void kernel_launch(void* const* d_in, const int* in_sizes, int n_in,
                              void* d_out, int out_size, void* d_ws, size_t ws_size,
                              hipStream_t stream) {
  const float* p1     = (const float*)d_in[0];
  const float* p3     = (const float*)d_in[1];
  const float* basis  = (const float*)d_in[2];
  const float* diff   = (const float*)d_in[3];
  const float* pi_W0  = (const float*)d_in[4];
  const float* pi_W1  = (const float*)d_in[5];
  const float* ii_W   = (const float*)d_in[6];
  const float* ipp_W  = (const float*)d_in[7];
  const float* ipp_b  = (const float*)d_in[8];
  const float* epi_W  = (const float*)d_in[9];
  const float* epp_W  = (const float*)d_in[10];
  const float* gpp_W  = (const float*)d_in[11];
  const float* gpp_b  = (const float*)d_in[12];
  const int*   pair_i = (const int*)d_in[13];
  const int*   pair_j = (const int*)d_in[14];

  const int N = in_sizes[0] / FDIM;   // 10000
  const int E = in_sizes[13];         // 320000

  float* out = (float*)d_out;
  float* out_p1t1 = out;                          // N*1*64
  float* out_p3t1 = out + (size_t)N * FDIM;       // N*3*64
  float* out_i1   = out + (size_t)N * 4 * FDIM;   // E*1*64
  float* out_ix   = out_i1 + (size_t)E * FDIM;    // E*3*64

  // Workspace: [p1acc f32 N*64][p3acc f32 N*192][f16 transposed weights]
  char* ws = (char*)d_ws;
  float* p1acc = (float*)ws;
  float* p3acc = (float*)(ws + (size_t)N * FDIM * sizeof(float));
  _Float16* hb = (_Float16*)(ws + (size_t)N * 4 * FDIM * sizeof(float));
  _Float16* w0T  = hb;                  // 64*64
  _Float16* w1T  = hb + 4096;           // 512*64
  _Float16* iiT  = hb + 4096 + 32768;
  _Float16* epiT = iiT + 4096;
  _Float16* ippT = epiT + 4096;
  _Float16* eppT = ippT + 4096;
  _Float16* gppT = eppT + 4096;

  // Zero segment-sum accumulators (p1acc and p3acc are contiguous)
  int accN = N * 4 * FDIM;
  zero_f32_kernel<<<(accN + 255) / 256, 256, 0, stream>>>(p1acc, accN);

  // Transpose weights to f16 W^T (N x K) for B-fragment loads
  transpose_to_f16_kernel<<<(4096 + 255) / 256, 256, 0, stream>>>(pi_W0, w0T, FDIM, FDIM);
  transpose_to_f16_kernel<<<(32768 + 255) / 256, 256, 0, stream>>>(pi_W1, w1T, FDIM, 512);
  transpose_to_f16_kernel<<<(4096 + 255) / 256, 256, 0, stream>>>(ii_W,  iiT,  FDIM, FDIM);
  transpose_to_f16_kernel<<<(4096 + 255) / 256, 256, 0, stream>>>(epi_W, epiT, FDIM, FDIM);
  transpose_to_f16_kernel<<<(4096 + 255) / 256, 256, 0, stream>>>(ipp_W, ippT, FDIM, FDIM);
  transpose_to_f16_kernel<<<(4096 + 255) / 256, 256, 0, stream>>>(epp_W, eppT, FDIM, FDIM);
  transpose_to_f16_kernel<<<(4096 + 255) / 256, 256, 0, stream>>>(gpp_W, gppT, FDIM, FDIM);

  // Edge pass: 16 edges/wave, 4 waves/block
  int edgeTiles  = (E + 15) / 16;
  int edgeBlocks = (edgeTiles + 3) / 4;
  edge_kernel<<<edgeBlocks, 128, 0, stream>>>(p1, p3, basis, diff, pair_i, pair_j,
                                              w0T, w1T, iiT, epiT,
                                              out_i1, out_ix, p1acc, p3acc, E);

  // Node pass: 16 nodes/wave, 2 waves/block
  int nodeTiles  = (N + 15) / 16;
  int nodeBlocks = (nodeTiles + 1) / 2;
  node_kernel<<<nodeBlocks, 64, 0, stream>>>(p1acc, p3acc, ippT, ipp_b, eppT,
                                             gppT, gpp_b, out_p1t1, out_p3t1, N);
}
